// IntentClassifier_10977936408755
// MI455X (gfx1250) — compile-verified, matrix operand order checked by reading, not run on hardware
//
#include <hip/hip_runtime.h>
#include <hip/hip_bf16.h>
#include <math.h>

// ---- problem constants ----
constexpr int cV = 50000, cE = 300, cH = 512, cO = 64, cT = 21;
constexpr int cB = 256, cS = 128;
constexpr int cEp = 320;            // E padded to multiple of 32 (K dim for WMMA)
constexpr int cG = 3 * cH;          // 1536
constexpr int cH2 = 2 * cH;         // 1024
constexpr int cSB = cB * cS;        // 32768

typedef _Float16 f16x8  __attribute__((ext_vector_type(8)));
typedef _Float16 f16x16 __attribute__((ext_vector_type(16)));
typedef float    f32x8  __attribute__((ext_vector_type(8)));

// Load a 16x32 f16 fragment (A or B) for V_WMMA_F32_16X16X32_F16.
// ISA layout (16-bit A 16x32): lane = 16g + (row%16); per lane halves:
//   e in [0,8)  -> K = k0 + 8g + e
//   e in [8,16) -> K = k0 + 16 + 8g + (e-8)
__device__ __forceinline__ f16x16 load_frag(const _Float16* __restrict__ base,
                                            int ld, int row0, int k0, int lane) {
  int r = row0 + (lane & 15);
  int g = (lane >> 4) & 1;
  const _Float16* p = base + (size_t)r * ld + k0 + 8 * g;
  f16x8 lo = *(const f16x8*)(p);
  f16x8 hi = *(const f16x8*)(p + 16);
  return __builtin_shufflevector(lo, hi, 0,1,2,3,4,5,6,7,8,9,10,11,12,13,14,15);
}

// Register-blocked WMMA GEMM:
//   C[dir] = A[dir] (MxK, row-major f16) * B[dir]^T (NxK rows, f16) + bias[dir]
// One wave computes a 32x64 macro-tile: 2 A-frags x 4 B-frags -> 8 WMMAs / K-step.
// blockIdx = (N/64, M/32, dir)
__global__ __launch_bounds__(32)
void k_gemm(const _Float16* __restrict__ A, size_t sA,
            const _Float16* __restrict__ Bw, size_t sB,
            const float* __restrict__ bias0, const float* __restrict__ bias1,
            float* __restrict__ C, size_t sC, int N, int K) {
  int dir = blockIdx.z;
  A  += (size_t)dir * sA;
  Bw += (size_t)dir * sB;
  C  += (size_t)dir * sC;
  const float* bias = dir ? bias1 : bias0;

  int n0 = blockIdx.x * 64;
  int m0 = blockIdx.y * 32;
  int lane = threadIdx.x & 31;
  int g = lane >> 4;
  int nl = lane & 15;

  f32x8 acc[2][4];
#pragma unroll
  for (int ni = 0; ni < 4; ++ni) {
    float bv = bias[n0 + ni * 16 + nl];
#pragma unroll
    for (int mi = 0; mi < 2; ++mi)
#pragma unroll
      for (int r = 0; r < 8; ++r) acc[mi][ni][r] = bv;
  }

  for (int k0 = 0; k0 < K; k0 += 32) {
    f16x16 a0 = load_frag(A, K, m0,      k0, lane);
    f16x16 a1 = load_frag(A, K, m0 + 16, k0, lane);
    f16x16 b0 = load_frag(Bw, K, n0,      k0, lane);
    f16x16 b1 = load_frag(Bw, K, n0 + 16, k0, lane);
    f16x16 b2 = load_frag(Bw, K, n0 + 32, k0, lane);
    f16x16 b3 = load_frag(Bw, K, n0 + 48, k0, lane);
    acc[0][0] = __builtin_amdgcn_wmma_f32_16x16x32_f16(false, a0, false, b0, (short)0, acc[0][0], false, false);
    acc[0][1] = __builtin_amdgcn_wmma_f32_16x16x32_f16(false, a0, false, b1, (short)0, acc[0][1], false, false);
    acc[0][2] = __builtin_amdgcn_wmma_f32_16x16x32_f16(false, a0, false, b2, (short)0, acc[0][2], false, false);
    acc[0][3] = __builtin_amdgcn_wmma_f32_16x16x32_f16(false, a0, false, b3, (short)0, acc[0][3], false, false);
    acc[1][0] = __builtin_amdgcn_wmma_f32_16x16x32_f16(false, a1, false, b0, (short)0, acc[1][0], false, false);
    acc[1][1] = __builtin_amdgcn_wmma_f32_16x16x32_f16(false, a1, false, b1, (short)0, acc[1][1], false, false);
    acc[1][2] = __builtin_amdgcn_wmma_f32_16x16x32_f16(false, a1, false, b2, (short)0, acc[1][2], false, false);
    acc[1][3] = __builtin_amdgcn_wmma_f32_16x16x32_f16(false, a1, false, b3, (short)0, acc[1][3], false, false);
  }

#pragma unroll
  for (int mi = 0; mi < 2; ++mi)
#pragma unroll
    for (int ni = 0; ni < 4; ++ni) {
#pragma unroll
      for (int r = 0; r < 8; ++r) {
        int m = m0 + mi * 16 + r + 8 * g;
        C[(size_t)m * N + n0 + ni * 16 + nl] = acc[mi][ni][r];
      }
    }
}

// ---- conversion / gather kernels ----
__global__ void k_conv_wih(const float* __restrict__ wf, const float* __restrict__ wb,
                           _Float16* __restrict__ out) {
  int idx = blockIdx.x * blockDim.x + threadIdx.x;
  if (idx >= 2 * cG * cEp) return;
  int dir = idx / (cG * cEp);
  int rem = idx - dir * (cG * cEp);
  int row = rem / cEp;
  int col = rem - row * cEp;
  const float* w = dir ? wb : wf;
  out[idx] = (col < cE) ? (_Float16)w[row * cE + col] : (_Float16)0.f;
}

__global__ void k_conv_whh(const float* __restrict__ wf, const float* __restrict__ wb,
                           _Float16* __restrict__ out) {
  int idx = blockIdx.x * blockDim.x + threadIdx.x;
  if (idx >= 2 * cG * cH) return;
  int dir = idx / (cG * cH);
  int rem = idx - dir * (cG * cH);
  const float* w = dir ? wb : wf;
  out[idx] = (_Float16)w[rem];
}

// xe16[(s*B + b)][e] = emb[x[b][s]][e], K-padded with zeros
__global__ void k_embed(const int* __restrict__ x, const float* __restrict__ emb,
                        _Float16* __restrict__ out) {
  size_t idx = (size_t)blockIdx.x * blockDim.x + threadIdx.x;
  if (idx >= (size_t)cSB * cEp) return;
  int m = (int)(idx / cEp);
  int e = (int)(idx - (size_t)m * cEp);
  int s = m / cB;
  int b = m - s * cB;
  int tok = x[b * cS + s];
  out[idx] = (e < cE) ? (_Float16)emb[(size_t)tok * cE + e] : (_Float16)0.f;
}

__global__ void k_zero_h(float* __restrict__ h, _Float16* __restrict__ h16) {
  int idx = blockIdx.x * blockDim.x + threadIdx.x;
  if (idx >= 2 * cB * cH) return;
  h[idx] = 0.f;
  h16[idx] = (_Float16)0.f;
}

// GRU gate update for step t, both directions.
__global__ void k_gate(const float* __restrict__ xp, const float* __restrict__ hp,
                       float* __restrict__ h, _Float16* __restrict__ h16,
                       float* __restrict__ gru, int t) {
  int idx = blockIdx.x * blockDim.x + threadIdx.x;
  if (idx >= 2 * cB * cH) return;
  int dir = idx / (cB * cH);
  int rem = idx - dir * (cB * cH);
  int b = rem / cH;
  int j = rem - b * cH;
  int s = (dir == 0) ? t : (cS - 1 - t);

  const float* xrow = xp + (size_t)dir * cSB * cG + ((size_t)s * cB + b) * cG;
  const float* hrow = hp + (size_t)dir * cB * cG + (size_t)b * cG;
  float xr = xrow[j], xz = xrow[cH + j], xn = xrow[2 * cH + j];
  float hr = hrow[j], hz = hrow[cH + j], hn = hrow[2 * cH + j];

  float r = 1.f / (1.f + expf(-(xr + hr)));
  float z = 1.f / (1.f + expf(-(xz + hz)));
  float n = tanhf(xn + r * hn);

  size_t hidx = (size_t)dir * cB * cH + rem;
  float hold = h[hidx];
  float hv = (1.f - z) * n + z * hold;
  h[hidx] = hv;
  h16[hidx] = (_Float16)hv;
  gru[((size_t)b * cS + s) * cH2 + dir * cH + j] = hv;
}

// scores[b][s] = gru[b][s] . attn_w + attn_b
__global__ void k_attn(const float* __restrict__ gru, const float* __restrict__ aw,
                       const float* __restrict__ ab, float* __restrict__ scores) {
  int idx = blockIdx.x * blockDim.x + threadIdx.x;
  if (idx >= cSB) return;
  const float* grow = gru + (size_t)idx * cH2;
  float acc = ab[0];
  for (int k = 0; k < cH2; ++k) acc += grow[k] * aw[k];
  scores[idx] = acc;
}

__global__ __launch_bounds__(128)
void k_softmax(float* __restrict__ scores) {
  __shared__ float buf[cS];
  int b = blockIdx.x, s = threadIdx.x;
  float v = scores[b * cS + s];
  buf[s] = v; __syncthreads();
  for (int st = 64; st > 0; st >>= 1) { if (s < st) buf[s] = fmaxf(buf[s], buf[s + st]); __syncthreads(); }
  float m = buf[0]; __syncthreads();
  float e = expf(v - m);
  buf[s] = e; __syncthreads();
  for (int st = 64; st > 0; st >>= 1) { if (s < st) buf[s] += buf[s + st]; __syncthreads(); }
  scores[b * cS + s] = e / buf[0];
}

__global__ void k_ctx(const float* __restrict__ w, const float* __restrict__ gru,
                      float* __restrict__ ctx) {
  int idx = blockIdx.x * blockDim.x + threadIdx.x;
  if (idx >= cB * cH2) return;
  int b = idx / cH2;
  int k = idx - b * cH2;
  float acc = 0.f;
  for (int s = 0; s < cS; ++s)
    acc += w[b * cS + s] * gru[((size_t)b * cS + s) * cH2 + k];
  ctx[idx] = acc;
}

__global__ void k_intent(const float* __restrict__ ctx, const float* __restrict__ fw,
                         const float* __restrict__ fb, float* __restrict__ out) {
  int idx = blockIdx.x * blockDim.x + threadIdx.x;
  if (idx >= cB * cO) return;
  int b = idx / cO;
  int o = idx - b * cO;
  float acc = fb[o];
  const float* c = ctx + (size_t)b * cH2;
  const float* w = fw + (size_t)o * cH2;
  for (int k = 0; k < cH2; ++k) acc += c[k] * w[k];
  out[idx] = acc;
}

__global__ void k_emis(const float* __restrict__ gru, const float* __restrict__ nw,
                       const float* __restrict__ nb, float* __restrict__ emis) {
  int idx = blockIdx.x * blockDim.x + threadIdx.x;
  if (idx >= cSB * cT) return;
  int m = idx / cT;
  int tg = idx - m * cT;
  float acc = nb[tg];
  const float* grow = gru + (size_t)m * cH2;
  const float* w = nw + (size_t)tg * cH2;
  for (int k = 0; k < cH2; ++k) acc += grow[k] * w[k];
  emis[idx] = acc;
}

// One wave32 block per batch row: CRF forward scan + logZ + gold-path score.
__global__ __launch_bounds__(32)
void k_crf(const float* __restrict__ emis, const int* __restrict__ tags,
           const unsigned char* __restrict__ mask,
           const float* __restrict__ start, const float* __restrict__ endv,
           const float* __restrict__ trans, float* __restrict__ nll_b) {
  __shared__ float tr[cT * cT];
  __shared__ float sc[32];
  int b = blockIdx.x;
  int tid = threadIdx.x;
  for (int i = tid; i < cT * cT; i += 32) tr[i] = trans[i];
  __syncthreads();

  const float NEG = -1e30f;
  const float* em = emis + (size_t)b * cS * cT;
  float score = NEG;
  if (tid < cT) score = start[tid] + em[tid];

  for (int t = 1; t < cS; ++t) {
    sc[tid] = score;
    __syncthreads();
    float ns = score;
    if (tid < cT) {
      float m = NEG;
      for (int i = 0; i < cT; ++i) m = fmaxf(m, sc[i] + tr[i * cT + tid]);
      float ssum = 0.f;
      for (int i = 0; i < cT; ++i) ssum += expf(sc[i] + tr[i * cT + tid] - m);
      ns = m + logf(ssum) + em[t * cT + tid];
    }
    if (tid < cT && mask[b * cS + t]) score = ns;
    __syncthreads();
  }

  sc[tid] = (tid < cT) ? score + endv[tid] : NEG;
  __syncthreads();
  if (tid == 0) {
    float m = NEG;
    for (int i = 0; i < cT; ++i) m = fmaxf(m, sc[i]);
    float ssum = 0.f;
    for (int i = 0; i < cT; ++i) ssum += expf(sc[i] - m);
    float logZ = m + logf(ssum);

    const int* tg = tags + b * cS;
    int prev = tg[0], last = tg[0];
    float num = start[prev] + em[prev];
    for (int t = 1; t < cS; ++t) {
      int cur = tg[t];
      float mf = mask[b * cS + t] ? 1.f : 0.f;
      num += mf * (em[t * cT + cur] + tr[prev * cT + cur]);
      if (mask[b * cS + t]) last = cur;
      prev = cur;
    }
    num += endv[last];
    nll_b[b] = num - logZ;
  }
}

__global__ __launch_bounds__(256)
void k_loss(const float* __restrict__ nll_b, float* __restrict__ out) {
  __shared__ float s[cB];
  int tid = threadIdx.x;
  s[tid] = nll_b[tid];
  __syncthreads();
  for (int st = 128; st > 0; st >>= 1) { if (tid < st) s[tid] += s[tid + st]; __syncthreads(); }
  if (tid == 0) out[0] = -(s[0] / (float)cB);
}

extern "C" void kernel_launch(void* const* d_in, const int* in_sizes, int n_in,
                              void* d_out, int out_size, void* d_ws, size_t ws_size,
                              hipStream_t stream) {
  const int*   x        = (const int*)d_in[0];
  const int*   ner_tags = (const int*)d_in[1];
  const unsigned char* mask = (const unsigned char*)d_in[2];
  const float* emb      = (const float*)d_in[3];
  const float* W_ih_f   = (const float*)d_in[4];
  const float* W_hh_f   = (const float*)d_in[5];
  const float* b_ih_f   = (const float*)d_in[6];
  const float* b_hh_f   = (const float*)d_in[7];
  const float* W_ih_b   = (const float*)d_in[8];
  const float* W_hh_b   = (const float*)d_in[9];
  const float* b_ih_b   = (const float*)d_in[10];
  const float* b_hh_b   = (const float*)d_in[11];
  const float* attn_w   = (const float*)d_in[12];
  const float* attn_b   = (const float*)d_in[13];
  const float* fc_w     = (const float*)d_in[14];
  const float* fc_b     = (const float*)d_in[15];
  const float* ner_w    = (const float*)d_in[16];
  const float* ner_b    = (const float*)d_in[17];
  const float* crf_start= (const float*)d_in[18];
  const float* crf_end  = (const float*)d_in[19];
  const float* crf_trans= (const float*)d_in[20];

  char* ws = (char*)d_ws;
  size_t off = 0;
  auto alloc = [&](size_t bytes) -> void* {
    void* p = (void*)(ws + off);
    off += (bytes + 255) & ~(size_t)255;
    return p;
  };

  _Float16* xe16  = (_Float16*)alloc(sizeof(_Float16) * (size_t)cSB * cEp);
  _Float16* wih16 = (_Float16*)alloc(sizeof(_Float16) * 2 * (size_t)cG * cEp);
  _Float16* whh16 = (_Float16*)alloc(sizeof(_Float16) * 2 * (size_t)cG * cH);
  float*    xp    = (float*)alloc(sizeof(float) * 2 * (size_t)cSB * cG);
  float*    hbuf  = (float*)alloc(sizeof(float) * 2 * (size_t)cB * cH);
  _Float16* h16   = (_Float16*)alloc(sizeof(_Float16) * 2 * (size_t)cB * cH);
  float*    hp    = (float*)alloc(sizeof(float) * 2 * (size_t)cB * cG);
  float*    gru   = (float*)alloc(sizeof(float) * (size_t)cSB * cH2);
  float*    scores= (float*)alloc(sizeof(float) * (size_t)cSB);
  float*    ctx   = (float*)alloc(sizeof(float) * (size_t)cB * cH2);
  float*    emis  = (float*)alloc(sizeof(float) * (size_t)cSB * cT);
  float*    nllb  = (float*)alloc(sizeof(float) * cB);

  float* out = (float*)d_out;

  // weight / activation conversions
  {
    int n = 2 * cG * cEp;
    k_conv_wih<<<(n + 255) / 256, 256, 0, stream>>>(W_ih_f, W_ih_b, wih16);
  }
  {
    int n = 2 * cG * cH;
    k_conv_whh<<<(n + 255) / 256, 256, 0, stream>>>(W_hh_f, W_hh_b, whh16);
  }
  {
    size_t n = (size_t)cSB * cEp;
    k_embed<<<(unsigned)((n + 255) / 256), 256, 0, stream>>>(x, emb, xe16);
  }

  // input projections (WMMA GEMM): xp[dir] = xe16 @ Wih[dir]^T + b_ih[dir]
  k_gemm<<<dim3(cG / 64, cSB / 32, 2), 32, 0, stream>>>(
      xe16, 0, wih16, (size_t)cG * cEp, b_ih_f, b_ih_b,
      xp, (size_t)cSB * cG, cG, cEp);

  k_zero_h<<<(2 * cB * cH + 255) / 256, 256, 0, stream>>>(hbuf, h16);

  // sequential GRU recurrence: hp = h @ Whh^T + b_hh, then gates
  for (int t = 0; t < cS; ++t) {
    k_gemm<<<dim3(cG / 64, cB / 32, 2), 32, 0, stream>>>(
        h16, (size_t)cB * cH, whh16, (size_t)cG * cH, b_hh_f, b_hh_b,
        hp, (size_t)cB * cG, cG, cH);
    k_gate<<<(2 * cB * cH + 255) / 256, 256, 0, stream>>>(xp, hp, hbuf, h16, gru, t);
  }

  // attention + intent head
  k_attn<<<(cSB + 255) / 256, 256, 0, stream>>>(gru, attn_w, attn_b, scores);
  k_softmax<<<cB, 128, 0, stream>>>(scores);
  k_ctx<<<(cB * cH2 + 255) / 256, 256, 0, stream>>>(scores, gru, ctx);
  k_intent<<<(cB * cO + 255) / 256, 256, 0, stream>>>(ctx, fc_w, fc_b, out);

  // NER emissions + CRF NLL
  k_emis<<<(cSB * cT + 255) / 256, 256, 0, stream>>>(gru, ner_w, ner_b, emis);
  k_crf<<<cB, 32, 0, stream>>>(emis, ner_tags, mask, crf_start, crf_end, crf_trans, nllb);
  k_loss<<<1, 256, 0, stream>>>(nllb, out + cB * cO);

  (void)in_sizes; (void)n_in; (void)out_size; (void)ws_size;
}
